// TernaryLinear_18245021073626
// MI455X (gfx1250) — matrix operand order repaired and checked by reading, not compiled, not run on hardware
//
#include <hip/hip_runtime.h>

typedef __attribute__((ext_vector_type(16))) _Float16 v16h;
typedef __attribute__((ext_vector_type(8)))  _Float16 v8h;
typedef __attribute__((ext_vector_type(4))) _Float16 v4h;
typedef __attribute__((ext_vector_type(8)))  float    v8f;
typedef __attribute__((ext_vector_type(4)))  float    v4f;

constexpr int KDIM = 4096;   // in_features
constexpr int NDIM = 4096;   // out_features
constexpr int BM   = 128;    // block tile M
constexpr int BN   = 256;    // block tile N
constexpr int BK   = 32;     // K chunk
constexpr int LDSW = BK + 8; // f16 elems; 80B row stride -> 16B aligned, bank-friendly

// 8 waves: 2 (M) x 4 (N); wave tile 64x64 = 4x4 subtiles of 16x16.
// Double-buffered LDS: one barrier per K-step, staging overlaps WMMA phase.

__global__ __launch_bounds__(256)
void ternary_wmma_gemm(const float* __restrict__ X,
                       const float* __restrict__ Wt,   // (N, K) row-major
                       const float* __restrict__ bias,
                       float* __restrict__ C)
{
    __shared__ _Float16 As[2][BM * LDSW];   // 2 x 10,240 B
    __shared__ _Float16 Bs[2][BN * LDSW];   // 2 x 20,480 B

    const int t    = threadIdx.x;
    const int lane = t & 31;
    const int wave = t >> 5;      // 0..7
    const int wm   = wave >> 2;   // 0..1  -> 64 rows of M each
    const int wn   = wave & 3;    // 0..3  -> 64 cols of N each
    const int half = lane >> 4;   // 0/1 (wave halves per ISA fragment layout)
    const int l16  = lane & 15;

    const int m0 = blockIdx.y * BM;
    const int n0 = blockIdx.x * BN;

    // Global staging: 8 threads cover one 32-float row segment (128B).
    const int lr = t >> 3;        // 0..31
    const int lc = (t & 7) * 4;   // 0,4,...,28

    const float* Xp = X  + (size_t)(m0 + lr) * KDIM + lc;
    const float* Wp = Wt + (size_t)(n0 + lr) * KDIM + lc;

    v4f areg[4];                  // A tile: 128 rows -> 4 rows/thread
    v4f breg[8];                  // B tile: 256 rows -> 8 rows/thread

    // ---- Prologue: stage tile 0 into LDS buffer 0 ----
#pragma unroll
    for (int i = 0; i < 4; ++i)
        areg[i] = *(const v4f*)(Xp + (size_t)(32 * i) * KDIM);
#pragma unroll
    for (int i = 0; i < 8; ++i)
        breg[i] = *(const v4f*)(Wp + (size_t)(32 * i) * KDIM);
#pragma unroll
    for (int i = 0; i < 4; ++i)
        *(v4h*)(&As[0][(lr + 32 * i) * LDSW + lc]) = __builtin_convertvector(areg[i], v4h);
#pragma unroll
    for (int i = 0; i < 8; ++i)
        *(v4h*)(&Bs[0][(lr + 32 * i) * LDSW + lc]) = __builtin_convertvector(breg[i], v4h);
    __syncthreads();

    v8f acc[4][4] = {};
    int buf = 0;

    for (int k0 = 0; k0 < KDIM; k0 += BK) {
        const bool has_next = (k0 + BK < KDIM);

        // Issue next K-tile's global loads; they land while the WMMAs run.
        if (has_next) {
#pragma unroll
            for (int i = 0; i < 4; ++i)
                areg[i] = *(const v4f*)(Xp + (size_t)(32 * i) * KDIM + (k0 + BK));
#pragma unroll
            for (int i = 0; i < 8; ++i)
                breg[i] = *(const v4f*)(Wp + (size_t)(32 * i) * KDIM + (k0 + BK));
        }

        // A fragments (16x32 f16): half 0 -> K 0..7 & 16..23, half 1 -> K 8..15 & 24..31
        v16h Af[4];
#pragma unroll
        for (int mi = 0; mi < 4; ++mi) {
            const _Float16* p = &As[buf][(wm * 64 + mi * 16 + l16) * LDSW + half * 8];
            v8h lo = *(const v8h*)p;
            v8h hi = *(const v8h*)(p + 16);
            Af[mi] = __builtin_shufflevector(lo, hi, 0, 1, 2, 3, 4, 5, 6, 7,
                                                     8, 9, 10, 11, 12, 13, 14, 15);
        }

        // B fragments (32x16 f16): N = lane&15, half 0 -> K 0..7 & 16..23, half 1 -> K 8..15 & 24..31
        v16h Bf[4];
#pragma unroll
        for (int ni = 0; ni < 4; ++ni) {
            const _Float16* p = &Bs[buf][(wn * 64 + ni * 16 + l16) * LDSW + half * 8];
            v8h lo = *(const v8h*)p;
            v8h hi = *(const v8h*)(p + 16);
            Bf[ni] = __builtin_shufflevector(lo, hi, 0, 1, 2, 3, 4, 5, 6, 7,
                                                     8, 9, 10, 11, 12, 13, 14, 15);
        }

        // 16 WMMAs per wave per K-step (64x64 wave tile).
#pragma unroll
        for (int mi = 0; mi < 4; ++mi)
#pragma unroll
            for (int ni = 0; ni < 4; ++ni)
                acc[mi][ni] = __builtin_amdgcn_wmma_f32_16x16x32_f16(
                    /*neg_a=*/false, Af[mi], /*neg_b=*/false, Bf[ni],
                    /*c_mod=*/(short)0, acc[mi][ni],
                    /*reuse_a=*/false, /*reuse_b=*/false);

        // Stage tile k+1 into the other buffer; overlaps the XDL pipe above.
        if (has_next) {
            const int nb = buf ^ 1;
#pragma unroll
            for (int i = 0; i < 4; ++i)
                *(v4h*)(&As[nb][(lr + 32 * i) * LDSW + lc]) = __builtin_convertvector(areg[i], v4h);
#pragma unroll
            for (int i = 0; i < 8; ++i)
                *(v4h*)(&Bs[nb][(lr + 32 * i) * LDSW + lc]) = __builtin_convertvector(breg[i], v4h);
            __syncthreads();   // uniform condition: safe
        }
        buf ^= 1;
    }

    // Epilogue: C layout — VGPR j holds M = j (lanes 0-15) / j+8 (lanes 16-31), N = lane&15.
#pragma unroll
    for (int ni = 0; ni < 4; ++ni) {
        const int col = n0 + wn * 64 + ni * 16 + l16;
        const float bv = bias[col];
#pragma unroll
        for (int mi = 0; mi < 4; ++mi) {
            const int rbase = m0 + wm * 64 + mi * 16 + half * 8;
#pragma unroll
            for (int j = 0; j < 8; ++j)
                C[(size_t)(rbase + j) * NDIM + col] = acc[mi][ni][j] + bv;
        }
    }
}

extern "C" void kernel_launch(void* const* d_in, const int* in_sizes, int n_in,
                              void* d_out, int out_size, void* d_ws, size_t ws_size,
                              hipStream_t stream) {
    (void)n_in; (void)out_size; (void)d_ws; (void)ws_size;
    const float* X    = (const float*)d_in[0];
    const float* Wt   = (const float*)d_in[1];
    const float* bias = (const float*)d_in[2];
    float* C          = (float*)d_out;

    const int M = in_sizes[0] / KDIM;         // 8192
    dim3 grid(NDIM / BN, M / BM);             // (16, 64)
    ternary_wmma_gemm<<<grid, 256, 0, stream>>>(X, Wt, bias, C);
}
